// PrototypeHead_73598559584741
// MI455X (gfx1250) — matrix operand order repaired and checked by reading, not compiled, not run on hardware
//
#include <hip/hip_runtime.h>
#include <math.h>

// ---------------------------------------------------------------------------
// PrototypeHead loss on MI455X (gfx1250, wave32, WMMA)
//   N = 32768 rows, D = 512 features, K = 512 prototypes
// ---------------------------------------------------------------------------

#define DD        512            // feature dim == prototype count
#define INV_EPS   20.0f          // 1/SINKHORN_EPS
#define INV_TAU   (1.0f/0.3f)
#define SK_ITERS  20

typedef __bf16 bf16;
typedef __attribute__((ext_vector_type(16))) bf16  v16bf;
typedef __attribute__((ext_vector_type(8)))  float v8f;

// ---------------- wave32 reductions ----------------------------------------
__device__ __forceinline__ float wave_sum(float x) {
#pragma unroll
  for (int o = 16; o > 0; o >>= 1) x += __shfl_xor(x, o, 32);
  return x;
}
__device__ __forceinline__ float wave_max(float x) {
#pragma unroll
  for (int o = 16; o > 0; o >>= 1) x = fmaxf(x, __shfl_xor(x, o, 32));
  return x;
}

// ---------------- row inverse L2 norm (one wave per 512-wide row) ----------
__global__ __launch_bounds__(256)
void invnorm_kernel(const float* __restrict__ x, float* __restrict__ inv, int rows) {
  const int wave = threadIdx.x >> 5, lane = threadIdx.x & 31;
  const int r = blockIdx.x * 8 + wave;
  if (r >= rows) return;
  const float* row = x + (size_t)r * DD;
  float s = 0.f;
#pragma unroll
  for (int j = 0; j < 16; ++j) { float v = row[lane + 32 * j]; s += v * v; }
  s = wave_sum(s);
  if (lane == 0) inv[r] = 1.0f / fmaxf(sqrtf(s), 1e-12f);
}

// ---------------- init: c = 1, p = 0 ---------------------------------------
__global__ void init_kernel(float* c1, float* c2, float* p1, float* p2, int N) {
  const int i = blockIdx.x * blockDim.x + threadIdx.x;
  if (i < N) { c1[i] = 1.f; c2[i] = 1.f; }
  if (i < DD) { p1[i] = 0.f; p2[i] = 0.f; }
}

// ---------------- one-shot bf16 hi/lo split of the prototype matrix --------
__global__ __launch_bounds__(256)
void wcvt_kernel(const float* __restrict__ w, bf16* __restrict__ whi,
                 bf16* __restrict__ wlo, int n) {
  const int i = blockIdx.x * blockDim.x + threadIdx.x;
  if (i >= n) return;
  const float f = w[i];
  const bf16 h = (bf16)f;
  whi[i] = h;
  wlo[i] = (bf16)(f - (float)h);
}

// ---------------- bf16 hi/lo split of 16 consecutive floats ----------------
__device__ __forceinline__ void cvt16(const float* __restrict__ p0,
                                      const float* __restrict__ p1,
                                      v16bf& hi, v16bf& lo) {
#pragma unroll
  for (int e = 0; e < 8; ++e) {
    float f = p0[e]; bf16 h = (bf16)f;
    hi[e] = h; lo[e] = (bf16)(f - (float)h);
  }
#pragma unroll
  for (int e = 0; e < 8; ++e) {
    float f = p1[e]; bf16 h = (bf16)f;
    hi[8 + e] = h; lo[8 + e] = (bf16)(f - (float)h);
  }
}

// ---------------- cosine GEMM: out[n,k] = (z_n . w_k) * invz[n] * invw[k] --
// Block = 256 threads = 8 waves arranged 2(row) x 4(col).
// Workgroup tile 64x256 (grid.y = 512/256 = 2 -> z read only twice total).
// Wave tile 32x64 = 2x4 WMMA 16x16 fragments.
// bf16 hi/lo split: 3 WMMAs per fragment per K-step (fp32-class accuracy).
// B fragments load pre-split bf16 directly as aligned 32B v16bf vectors.
__global__ __launch_bounds__(256)
void gemm_cos_kernel(const float* __restrict__ Z,
                     const bf16* __restrict__ Whi, const bf16* __restrict__ Wlo,
                     const float* __restrict__ invz, const float* __restrict__ invw,
                     float* __restrict__ out) {
  const int lane = threadIdx.x & 31;
  const int wave = threadIdx.x >> 5;
  const int wr = wave >> 2, wc = wave & 3;
  const int h = lane >> 4;          // lane half (K-group select)
  const int m = lane & 15;          // row/col within fragment

  const int rowBase = blockIdx.x * 64  + wr * 32;
  const int colBase = blockIdx.y * 256 + wc * 64;

  v8f acc[2][4];
#pragma unroll
  for (int i = 0; i < 2; ++i)
#pragma unroll
    for (int j = 0; j < 4; ++j) acc[i][j] = (v8f){0,0,0,0,0,0,0,0};

  for (int d0 = 0; d0 < DD; d0 += 32) {
    v16bf ahi[2], alo[2];
#pragma unroll
    for (int i = 0; i < 2; ++i) {
      // A k order used here (contiguous, same as B): half 0 -> K in {0..15}, half 1 -> {16..31}
      const float* ap = Z + (size_t)(rowBase + i * 16 + m) * DD + d0 + 16 * h;
      cvt16(ap, ap + 8, ahi[i], alo[i]);
    }
#pragma unroll
    for (int j = 0; j < 4; ++j) {
      // B k order used here (contiguous, same as A): lanes 0-15 hold K=0..15, lanes 16-31 hold K=16..31
      const size_t boff = (size_t)(colBase + j * 16 + m) * DD + d0 + 16 * h;
      const v16bf bhi = *(const v16bf*)(Whi + boff);   // 32B aligned vector load
      const v16bf blo = *(const v16bf*)(Wlo + boff);
#pragma unroll
      for (int i = 0; i < 2; ++i) {
        acc[i][j] = __builtin_amdgcn_wmma_f32_16x16x32_bf16(
            false, ahi[i], false, bhi, (short)0, acc[i][j], false, false);
        acc[i][j] = __builtin_amdgcn_wmma_f32_16x16x32_bf16(
            false, ahi[i], false, blo, (short)0, acc[i][j], false, false);
        acc[i][j] = __builtin_amdgcn_wmma_f32_16x16x32_bf16(
            false, alo[i], false, bhi, (short)0, acc[i][j], false, false);
      }
    }
  }

  // Epilogue: C 16x16 layout -> lane&15 = N, VGPR r = M - 8*half
#pragma unroll
  for (int j = 0; j < 4; ++j) {
    const int col = colBase + j * 16 + m;
    const float iw = invw[col];
#pragma unroll
    for (int i = 0; i < 2; ++i) {
#pragma unroll
      for (int r = 0; r < 8; ++r) {
        const int row = rowBase + i * 16 + 8 * h + r;
        out[(size_t)row * DD + col] = acc[i][j][r] * invz[row] * iw;
      }
    }
  }
}

// ---------------- Sinkhorn (diagonal-scaling form) -------------------------
// Q = diag(r) * exp(zc^T/eps) * diag(c).
// col-pass: p_k += sum_b exp(zc[b,k]/eps) * c_b   (block-partial + atomics)
__global__ __launch_bounds__(256)
void sk_colpass(const float* __restrict__ zc, const float* __restrict__ c,
                float* __restrict__ p, int rowsPerBlock) {
  const int t  = threadIdx.x;
  const int b0 = blockIdx.x * rowsPerBlock;
  float a0 = 0.f, a1 = 0.f;
  for (int b = b0; b < b0 + rowsPerBlock; ++b) {
    const float cb = c[b];
    const float* row = zc + (size_t)b * DD;
    __builtin_prefetch(row + 2 * DD, 0, 1);          // global_prefetch path
    a0 += __expf(row[t]       * INV_EPS) * cb;
    a1 += __expf(row[t + 256] * INV_EPS) * cb;
  }
  atomicAdd(&p[t], a0);
  atomicAdd(&p[t + 256], a1);
}

// r_k = 1/(K * p_k); re-zero p for next iteration
__global__ void sk_finalize_r(float* __restrict__ p, float* __restrict__ r) {
  const int k = blockIdx.x * blockDim.x + threadIdx.x;
  if (k < DD) { r[k] = 1.0f / ((float)DD * p[k]); p[k] = 0.f; }
}

// row-pass: c_b = 1/(B * sum_k exp(zc[b,k]/eps) * r_k)   (one wave per row)
__global__ __launch_bounds__(256)
void sk_rowpass(const float* __restrict__ zc, const float* __restrict__ r,
                float* __restrict__ c, int N) {
  const int wave = threadIdx.x >> 5, lane = threadIdx.x & 31;
  const int b = blockIdx.x * 8 + wave;
  if (b >= N) return;
  const float* row = zc + (size_t)b * DD;
  float s = 0.f;
#pragma unroll
  for (int j = 0; j < 16; ++j) {
    const int k = lane + 32 * j;
    s += __expf(row[k] * INV_EPS) * r[k];
  }
  s = wave_sum(s);
  if (lane == 0) c[b] = 1.0f / ((float)N * s);
}

// ---------------- fused cross-entropy loss ---------------------------------
// partial[blk] = sum over its rows b of  [ q1 . logsm2  +  q2 . logsm1 ]
__global__ __launch_bounds__(256)
void loss_kernel(const float* __restrict__ zc1, const float* __restrict__ zc2,
                 const float* __restrict__ r1, const float* __restrict__ c1,
                 const float* __restrict__ r2, const float* __restrict__ c2,
                 float* __restrict__ partial, int N) {
  __shared__ float acc_s;
  if (threadIdx.x == 0) acc_s = 0.f;
  __syncthreads();
  const int wave = threadIdx.x >> 5, lane = threadIdx.x & 31;
  float local = 0.f;
  for (int b = blockIdx.x * 8 + wave; b < N; b += gridDim.x * 8) {
    const float* row1 = zc1 + (size_t)b * DD;
    const float* row2 = zc2 + (size_t)b * DD;
    float x1[16], x2[16];
    float m1 = -3.402823466e38f, m2 = -3.402823466e38f;
#pragma unroll
    for (int j = 0; j < 16; ++j) {
      const int k = lane + 32 * j;
      x1[j] = row1[k]; x2[j] = row2[k];
      m1 = fmaxf(m1, x1[j] * INV_TAU);
      m2 = fmaxf(m2, x2[j] * INV_TAU);
    }
    m1 = wave_max(m1); m2 = wave_max(m2);
    float s1 = 0.f, s2 = 0.f;
#pragma unroll
    for (int j = 0; j < 16; ++j) {
      s1 += __expf(x1[j] * INV_TAU - m1);
      s2 += __expf(x2[j] * INV_TAU - m2);
    }
    s1 = wave_sum(s1); s2 = wave_sum(s2);
    const float lse1 = m1 + __logf(s1);
    const float lse2 = m2 + __logf(s2);
    const float cb1 = c1[b], cb2 = c2[b];
    float t = 0.f;
#pragma unroll
    for (int j = 0; j < 16; ++j) {
      const int k = lane + 32 * j;
      const float q1 = (float)N * r1[k] * __expf(x1[j] * INV_EPS) * cb1;
      const float q2 = (float)N * r2[k] * __expf(x2[j] * INV_EPS) * cb2;
      t += q1 * (x2[j] * INV_TAU - lse2) + q2 * (x1[j] * INV_TAU - lse1);
    }
    t = wave_sum(t);
    if (lane == 0) local += t;
  }
  if (lane == 0) atomicAdd(&acc_s, local);   // ds_add_f32
  __syncthreads();
  if (threadIdx.x == 0) partial[blockIdx.x] = acc_s;
}

// partial[0..n) -> out[0] = scale * sum
__global__ __launch_bounds__(512)
void reduce_kernel(const float* __restrict__ partial, float* __restrict__ out,
                   int n, float scale) {
  __shared__ float sm[512];
  sm[threadIdx.x] = (threadIdx.x < n) ? partial[threadIdx.x] : 0.f;
  __syncthreads();
#pragma unroll
  for (int o = 256; o > 0; o >>= 1) {
    if (threadIdx.x < o) sm[threadIdx.x] += sm[threadIdx.x + o];
    __syncthreads();
  }
  if (threadIdx.x == 0) out[0] = sm[0] * scale;
}

__global__ void fail_kernel(float* out) { if (threadIdx.x == 0) out[0] = 0.f; }

// ---------------------------------------------------------------------------
extern "C" void kernel_launch(void* const* d_in, const int* in_sizes, int n_in,
                              void* d_out, int out_size, void* d_ws, size_t ws_size,
                              hipStream_t stream) {
  const float* z1 = (const float*)d_in[0];
  const float* z2 = (const float*)d_in[1];
  const float* w  = (const float*)d_in[2];
  const int N = in_sizes[0] / DD;      // 32768
  float* outp = (float*)d_out;

  // ---- workspace carve-up (float units) ----
  float* ws = (float*)d_ws;
  size_t off = 0;
  float* zc1   = ws + off; off += (size_t)N * DD;
  float* zc2   = ws + off; off += (size_t)N * DD;
  float* invz1 = ws + off; off += N;
  float* invz2 = ws + off; off += N;
  float* invw  = ws + off; off += DD;
  float* c1    = ws + off; off += N;
  float* c2    = ws + off; off += N;
  float* r1    = ws + off; off += DD;
  float* r2    = ws + off; off += DD;
  float* p1    = ws + off; off += DD;
  float* p2    = ws + off; off += DD;
  float* part  = ws + off; off += 512;
  bf16*  whi   = (bf16*)(ws + off); off += (DD * DD) / 2;   // 512x512 bf16
  bf16*  wlo   = (bf16*)(ws + off); off += (DD * DD) / 2;
  if (ws_size < off * sizeof(float)) {      // not enough scratch: emit 0 deterministically
    fail_kernel<<<1, 64, 0, stream>>>(outp);
    return;
  }

  // ---- init + norms + prototype bf16 split ----
  init_kernel<<<(2 * N + 255) / 256, 256, 0, stream>>>(c1, c2, p1, p2, N);
  invnorm_kernel<<<(N + 7) / 8, 256, 0, stream>>>(z1, invz1, N);
  invnorm_kernel<<<(N + 7) / 8, 256, 0, stream>>>(z2, invz2, N);
  invnorm_kernel<<<(DD + 7) / 8, 256, 0, stream>>>(w, invw, DD);
  wcvt_kernel<<<(DD * DD + 255) / 256, 256, 0, stream>>>(w, whi, wlo, DD * DD);

  // ---- cosine-similarity GEMMs (WMMA bf16 hi/lo split) ----
  dim3 g(N / 64, DD / 256);
  gemm_cos_kernel<<<g, 256, 0, stream>>>(z1, whi, wlo, invz1, invw, zc1);
  gemm_cos_kernel<<<g, 256, 0, stream>>>(z2, whi, wlo, invz2, invw, zc2);

  // ---- Sinkhorn iterations (diagonal-scaling form) ----
  const int rowsPerBlock = 128;
  for (int m = 0; m < 2; ++m) {
    const float* zc = m ? zc2 : zc1;
    float* c = m ? c2 : c1;
    float* r = m ? r2 : r1;
    float* p = m ? p2 : p1;
    for (int it = 0; it < SK_ITERS; ++it) {
      sk_colpass<<<N / rowsPerBlock, 256, 0, stream>>>(zc, c, p, rowsPerBlock);
      sk_finalize_r<<<2, 256, 0, stream>>>(p, r);
      sk_rowpass<<<(N + 7) / 8, 256, 0, stream>>>(zc, r, c, N);
    }
  }

  // ---- fused loss + reduction ----
  loss_kernel<<<512, 256, 0, stream>>>(zc1, zc2, r1, c1, r2, c2, part, N);
  reduce_kernel<<<1, 512, 0, stream>>>(part, outp, 512, -1.0f / (float)N);
}
